// BlockModelWrapper_17703855194733
// MI455X (gfx1250) — matrix operand order repaired and checked by reading, not compiled, not run on hardware
//
#include <hip/hip_runtime.h>

// ---------------- problem constants ----------------
#define DMODEL  1024
#define DINNER  2048
#define DSTATE  16
#define DTRANK  64
#define DCONV   4
#define BATCH   2
#define SEQ     2048
#define MROWS   (BATCH * SEQ)       // 4096 token rows

typedef __bf16 bf16_t;
typedef __attribute__((ext_vector_type(16))) __bf16 v16bf;
typedef __attribute__((ext_vector_type(8)))  __bf16 v8bf;
typedef __attribute__((ext_vector_type(8)))  float  v8f;

// ---------------- WMMA helpers (layouts per CDNA5 ISA 7.12.2) ----
__device__ __forceinline__ v8f wmma_bf16(v16bf a, v16bf b, v8f c) {
  // 8 args: (neg_a, A, neg_b, B, c_mod, C, reuse_a, reuse_b)
  return __builtin_amdgcn_wmma_f32_16x16x32_bf16(false, a, false, b,
                                                 (short)0, c, false, false);
}

// A tile: 16(M) x 32(K) bf16 from row-major A[m][k], lda = K.
// lanes 0-15 : row = lane,    K = k0+0..7  (h[0]) and k0+8..15 (h[1])
// lanes 16-31: row = lane-16, K = k0+16..23 (h[0]) and k0+24..31 (h[1])
__device__ __forceinline__ v16bf load_a_frag(const bf16_t* __restrict__ A,
                                             int lda, int m0, int k0, int lane) {
  int row   = m0 + (lane & 15);
  int kbase = k0 + ((lane >> 4) ? 16 : 0);
  const bf16_t* p = A + (size_t)row * lda + kbase;
  union { v16bf v; v8bf h[2]; } u;
  u.h[0] = *(const v8bf*)(p);
  u.h[1] = *(const v8bf*)(p + 8);
  return u.v;
}

// B tile: 32(K) x 16(N) sourced from row-major W[n][k] (so B = W^T), ldw = K.
// lane = column n (mod 16); lanes 0-15 hold K=k0..k0+15, lanes 16-31 K=k0+16..31.
__device__ __forceinline__ v16bf load_b_frag(const bf16_t* __restrict__ W,
                                             int ldw, int n0, int k0, int lane) {
  int n  = n0 + (lane & 15);
  int kk = k0 + ((lane >> 4) << 4);
  return *(const v16bf*)(W + (size_t)n * ldw + kk);   // 32B aligned
}

// C/D tile 16x16 f32: VGPR r -> (lanes 0-15: M=r, N=lane) (lanes 16-31: M=r+8).
template <bool OUT_BF16>
__device__ __forceinline__ void store_c_frag(void* __restrict__ Cout, int ldc,
                                             int m0, int n0, int lane, v8f acc) {
  int col   = n0 + (lane & 15);
  int rbase = m0 + ((lane >> 4) ? 8 : 0);
#pragma unroll
  for (int r = 0; r < 8; ++r) {
    size_t idx = (size_t)(rbase + r) * ldc + col;
    if (OUT_BF16) ((bf16_t*)Cout)[idx] = (bf16_t)acc[r];
    else          ((float*)Cout)[idx]  = acc[r];
  }
}

// ---------------- register-tiled bf16 WMMA GEMM -----------------------------
// C[M,N] = A[M,K] * W[N,K]^T.  One wave owns a (16*MT) x (16*NT) output block:
// 8 WMMAs per 6 fragment loads at MT=2,NT=4. Single-buffered on purpose —
// live VGPRs = 8*MT*NT (acc) + 8*(MT+NT) (frags) ~= 112, no scratch spills;
// latency is hidden by occupancy + explicit prefetch one K-tile ahead.
template <int MT, int NT, bool OUT_BF16>
__global__ void gemm_wmma(const bf16_t* __restrict__ A,
                          const bf16_t* __restrict__ W,
                          void* __restrict__ Cout, int M, int N, int K) {
  int lane = threadIdx.x & 31;
  int wave = threadIdx.x >> 5;
  int m0 = blockIdx.y * (16 * MT);
  int n0 = (blockIdx.x * (blockDim.x >> 5) + wave) * (16 * NT);
  if (n0 >= N || m0 >= M) return;   // wave-uniform: EXEC stays all-ones

  v8f acc[MT][NT];
#pragma unroll
  for (int i = 0; i < MT; ++i)
#pragma unroll
    for (int j = 0; j < NT; ++j) acc[i][j] = (v8f)0.f;

  for (int k0 = 0; k0 < K; k0 += 32) {
    if (k0 + 32 < K) {  // global_prefetch_b8: warm next K-tile of the A stream
      __builtin_prefetch(A + (size_t)(m0 + (lane & 15)) * K + k0 + 32, 0, 0);
    }
    v16bf af[MT];
#pragma unroll
    for (int i = 0; i < MT; ++i)
      af[i] = load_a_frag(A, K, m0 + 16 * i, k0, lane);
#pragma unroll
    for (int j = 0; j < NT; ++j) {
      v16bf bf = load_b_frag(W, K, n0 + 16 * j, k0, lane);
#pragma unroll
      for (int i = 0; i < MT; ++i)
        acc[i][j] = wmma_bf16(af[i], bf, acc[i][j]);
    }
  }
#pragma unroll
  for (int i = 0; i < MT; ++i)
#pragma unroll
    for (int j = 0; j < NT; ++j)
      store_c_frag<OUT_BF16>(Cout, N, m0 + 16 * i, n0 + 16 * j, lane, acc[i][j]);
}

// ---------------- fp32 -> bf16 convert --------------------------------------
__global__ void f2bf_kernel(const float* __restrict__ in,
                            bf16_t* __restrict__ out, size_t n) {
  size_t i = (size_t)blockIdx.x * blockDim.x + threadIdx.x;
  if (i < n) out[i] = (bf16_t)in[i];
}

// ---------------- fused residual add + RMSNorm -> bf16 ----------------------
__global__ void addnorm_kernel(const float* __restrict__ hs,
                               const float* __restrict__ res,
                               const float* __restrict__ w,
                               bf16_t* __restrict__ out) {
  int row = blockIdx.x;
  const float* a = hs  + (size_t)row * DMODEL;
  const float* b = res + (size_t)row * DMODEL;
  __shared__ float red[256];
  float s = 0.f;
  for (int i = threadIdx.x; i < DMODEL; i += blockDim.x) {
    float v = a[i] + b[i];
    s += v * v;
  }
  red[threadIdx.x] = s;
  __syncthreads();
  for (int off = blockDim.x >> 1; off > 0; off >>= 1) {
    if (threadIdx.x < off) red[threadIdx.x] += red[threadIdx.x + off];
    __syncthreads();
  }
  float inv = rsqrtf(red[0] / (float)DMODEL + 1e-5f);
  for (int i = threadIdx.x; i < DMODEL; i += blockDim.x) {
    float v = (a[i] + b[i]) * inv * w[i];
    out[(size_t)row * DMODEL + i] = (bf16_t)v;
  }
}

// ---------------- causal depthwise conv (4 taps) + SiLU ---------------------
__global__ void conv_silu_kernel(const bf16_t* __restrict__ xz,
                                 const float* __restrict__ cw,
                                 const float* __restrict__ cb,
                                 bf16_t* __restrict__ xout) {
  size_t gid = (size_t)blockIdx.x * blockDim.x + threadIdx.x;
  if (gid >= (size_t)MROWS * DINNER) return;
  int d = (int)(gid % DINNER);
  size_t bl = gid / DINNER;
  int l = (int)(bl % SEQ);
  size_t b = bl / SEQ;
  float acc = cb[d];
#pragma unroll
  for (int j = 0; j < DCONV; ++j) {
    int ls = l - (DCONV - 1) + j;
    if (ls >= 0)
      acc += cw[d * DCONV + j] *
             (float)xz[((size_t)b * SEQ + ls) * (2 * DINNER) + d];
  }
  float s = acc / (1.f + __expf(-acc));
  xout[gid] = (bf16_t)s;
}

// ---------------- split x_dbl into dt_low(bf16) / B / C ---------------------
__global__ void split_xdbl_kernel(const float* __restrict__ xdbl,
                                  bf16_t* __restrict__ dtlow,
                                  float* __restrict__ Bm,
                                  float* __restrict__ Cm) {
  int m = blockIdx.x;
  int t = threadIdx.x;           // 96 threads
  float v = xdbl[(size_t)m * (DTRANK + 2 * DSTATE) + t];
  if (t < DTRANK)               dtlow[(size_t)m * DTRANK + t] = (bf16_t)v;
  else if (t < DTRANK + DSTATE) Bm[(size_t)m * DSTATE + (t - DTRANK)] = v;
  else                          Cm[(size_t)m * DSTATE + (t - DTRANK - DSTATE)] = v;
}

// ---------------- dt = softplus(dt + bias) ----------------------------------
__global__ void dt_softplus_kernel(float* __restrict__ dt,
                                   const float* __restrict__ bias) {
  size_t gid = (size_t)blockIdx.x * blockDim.x + threadIdx.x;
  if (gid >= (size_t)MROWS * DINNER) return;
  float v = dt[gid] + bias[gid % DINNER];
  dt[gid] = (v > 20.f) ? v : log1pf(__expf(v));
}

// ---------------- selective scan (chunked, async-LDS staged B/C) ------------
// One thread owns one (b,d) channel: 16 states + 16 A coeffs in VGPRs.
// B/C chunk staged global->LDS with gfx1250 async copies (ASYNCcnt), no VGPR
// round-trip; 8 independent dt/x/z loads issued ahead of the recurrence.
#define SCAN_CH 8
__global__ void scan_kernel(const float* __restrict__ dt,
                            const bf16_t* __restrict__ xbf,
                            const float* __restrict__ Bm,
                            const float* __restrict__ Cm,
                            const float* __restrict__ A_log,
                            const float* __restrict__ Dp,
                            const bf16_t* __restrict__ xz,
                            bf16_t* __restrict__ ybf) {
  int d = blockIdx.x * blockDim.x + threadIdx.x;   // 0..DINNER-1
  int b = blockIdx.y;
  float Arow[DSTATE], st[DSTATE];
#pragma unroll
  for (int n = 0; n < DSTATE; ++n) {
    Arow[n] = -__expf(A_log[(size_t)d * DSTATE + n]);
    st[n] = 0.f;
  }
  float Dv = Dp[d];
  __shared__ float sB[SCAN_CH][DSTATE];
  __shared__ float sC[SCAN_CH][DSTATE];

  // per-thread source/dest for the async B/C stage (256 thr == 256 floats)
  const int i = threadIdx.x;
  const int cj = (i & 127) >> 4, cn = i & 15;
  float* lds_dst = (i < SCAN_CH * DSTATE) ? &sB[cj][cn] : &sC[cj][cn];
  const float* gbase = (i < SCAN_CH * DSTATE) ? Bm : Cm;
  unsigned lds_off = (unsigned)(size_t)lds_dst;   // flat low 32b == LDS offset

  for (int t0 = 0; t0 < SEQ; t0 += SCAN_CH) {
    {
      const float* gsrc = gbase + ((size_t)b * SEQ + t0 + cj) * DSTATE + cn;
      unsigned long long ga = (unsigned long long)(size_t)gsrc;
      asm volatile("global_load_async_to_lds_b32 %0, %1, off"
                   :: "v"(lds_off), "v"(ga) : "memory");
      asm volatile("s_wait_asynccnt 0" ::: "memory");
    }
    __syncthreads();

    float dtv[SCAN_CH], xv[SCAN_CH], zv[SCAN_CH];
#pragma unroll
    for (int j = 0; j < SCAN_CH; ++j) {   // independent loads, overlap latency
      size_t row = (size_t)b * SEQ + t0 + j;
      dtv[j] = dt[row * DINNER + d];
      xv[j]  = (float)xbf[row * DINNER + d];
      zv[j]  = (float)xz[row * (2 * DINNER) + DINNER + d];
    }
#pragma unroll
    for (int j = 0; j < SCAN_CH; ++j) {   // dependent recurrence from regs/LDS
      float dtx = dtv[j] * xv[j];
      float y = 0.f;
#pragma unroll
      for (int n = 0; n < DSTATE; ++n) {
        float dA = __expf(dtv[j] * Arow[n]);
        st[n] = dA * st[n] + dtx * sB[j][n];
        y += st[n] * sC[j][n];
      }
      float z = zv[j];
      float o = (y + xv[j] * Dv) * (z / (1.f + __expf(-z)));
      ybf[((size_t)b * SEQ + t0 + j) * DINNER + d] = (bf16_t)o;
    }
    __syncthreads();
  }
}

// ---------------- host-side orchestration -----------------------------------
extern "C" void kernel_launch(void* const* d_in, const int* in_sizes, int n_in,
                              void* d_out, int out_size, void* d_ws, size_t ws_size,
                              hipStream_t stream) {
  (void)in_sizes; (void)n_in; (void)out_size; (void)ws_size;
  const float* hidden   = (const float*)d_in[1];
  const float* resid    = (const float*)d_in[2];
  const float* norm_w   = (const float*)d_in[3];
  const float* in_proj  = (const float*)d_in[4];   // (4096,1024)
  const float* conv_w   = (const float*)d_in[5];   // (2048,1,4)
  const float* conv_b   = (const float*)d_in[6];
  const float* x_proj   = (const float*)d_in[7];   // (96,2048)
  const float* dt_proj  = (const float*)d_in[8];   // (2048,64)
  const float* dt_bias  = (const float*)d_in[9];
  const float* A_log    = (const float*)d_in[10];  // (2048,16)
  const float* Dp       = (const float*)d_in[11];
  const float* out_proj = (const float*)d_in[12];  // (1024,2048)
  float* out = (float*)d_out;

  char* base = (char*)d_ws;
  size_t off = 0;
  auto alloc = [&](size_t bytes) -> char* {
    char* p = base + off;
    off = (off + bytes + 255) & ~(size_t)255;
    return p;
  };

  bf16_t* w_in   = (bf16_t*)alloc((size_t)2 * DINNER * DMODEL * 2);           // 8 MB
  bf16_t* w_xp   = (bf16_t*)alloc((size_t)(DTRANK + 2 * DSTATE) * DINNER * 2);
  bf16_t* w_dt   = (bf16_t*)alloc((size_t)DINNER * DTRANK * 2);
  bf16_t* w_out  = (bf16_t*)alloc((size_t)DMODEL * DINNER * 2);               // 4 MB
  bf16_t* hbf    = (bf16_t*)alloc((size_t)MROWS * DMODEL * 2);                // 8 MB
  bf16_t* xzbf   = (bf16_t*)alloc((size_t)MROWS * 2 * DINNER * 2);            // 32 MB
  bf16_t* xbf    = (bf16_t*)alloc((size_t)MROWS * DINNER * 2);                // 16 MB
  float*  xdbl   = (float*) alloc((size_t)MROWS * (DTRANK + 2 * DSTATE) * 4); // 1.5 MB
  bf16_t* dtlow  = (bf16_t*)alloc((size_t)MROWS * DTRANK * 2);
  float*  Bmat   = (float*) alloc((size_t)MROWS * DSTATE * 4);
  float*  Cmat   = (float*) alloc((size_t)MROWS * DSTATE * 4);
  float*  dtf    = (float*) alloc((size_t)MROWS * DINNER * 4);                // 32 MB
  bf16_t* ybf    = (bf16_t*)alloc((size_t)MROWS * DINNER * 2);                // 16 MB

  // --- weight conversion to bf16 ---
  auto cvt = [&](const float* src, bf16_t* dst, size_t n) {
    f2bf_kernel<<<(unsigned)((n + 255) / 256), 256, 0, stream>>>(src, dst, n);
  };
  cvt(in_proj,  w_in,  (size_t)2 * DINNER * DMODEL);
  cvt(x_proj,   w_xp,  (size_t)(DTRANK + 2 * DSTATE) * DINNER);
  cvt(dt_proj,  w_dt,  (size_t)DINNER * DTRANK);
  cvt(out_proj, w_out, (size_t)DMODEL * DINNER);

  // --- residual add + RMSNorm ---
  addnorm_kernel<<<MROWS, 256, 0, stream>>>(hidden, resid, norm_w, hbf);

  // --- GEMM1: xz = h @ in_proj^T  (M=4096, N=4096, K=1024) -> bf16 ---
  gemm_wmma<2, 4, true><<<dim3((2 * DINNER) / 256, MROWS / 32), 128, 0, stream>>>(
      hbf, w_in, xzbf, MROWS, 2 * DINNER, DMODEL);

  // --- causal conv + SiLU on x half ---
  conv_silu_kernel<<<(unsigned)(((size_t)MROWS * DINNER + 255) / 256), 256, 0, stream>>>(
      xzbf, conv_w, conv_b, xbf);

  // --- GEMM2: x_dbl = x @ x_proj^T  (M=4096, N=96, K=2048) -> f32 ---
  gemm_wmma<2, 2, false><<<dim3(1, MROWS / 32), 96, 0, stream>>>(
      xbf, w_xp, xdbl, MROWS, DTRANK + 2 * DSTATE, DINNER);

  // --- split into dt_low / B / C ---
  split_xdbl_kernel<<<MROWS, DTRANK + 2 * DSTATE, 0, stream>>>(xdbl, dtlow, Bmat, Cmat);

  // --- GEMM3: dt = dt_low @ dt_proj^T  (M=4096, N=2048, K=64) -> f32 ---
  gemm_wmma<2, 4, false><<<dim3(DINNER / 256, MROWS / 32), 128, 0, stream>>>(
      dtlow, w_dt, dtf, MROWS, DINNER, DTRANK);
  dt_softplus_kernel<<<(unsigned)(((size_t)MROWS * DINNER + 255) / 256), 256, 0, stream>>>(
      dtf, dt_bias);

  // --- selective scan (fused with +x*D and *silu(z)) -> ybf ---
  scan_kernel<<<dim3(DINNER / 256, BATCH), 256, 0, stream>>>(
      dtf, xbf, Bmat, Cmat, A_log, Dp, xzbf, ybf);

  // --- GEMM4: out = y @ out_proj^T  (M=4096, N=1024, K=2048) -> f32 ---
  gemm_wmma<2, 4, false><<<dim3(DMODEL / 256, MROWS / 32), 128, 0, stream>>>(
      ybf, w_out, out, MROWS, DMODEL, DINNER);
}
